// SoftGroupEvidenceKofNNet_87531433492888
// MI455X (gfx1250) — matrix operand order repaired and checked by reading, not compiled, not run on hardware
//
#include <hip/hip_runtime.h>

#define BN 1024
#define RN 256
#define DN 512
#define GN 4
#define RDN (RN * DN)
#define LXS 516   // padded LDS row stride (dwords): 516 % 64 == 4 -> distinct banks per row

typedef __attribute__((ext_vector_type(16))) _Float16 v16h;
typedef __attribute__((ext_vector_type(8)))  float    v8f;
typedef __attribute__((address_space(3)))    void     lds_void;

// gfx1250 V_TANH_F32 (confirmed in round-1 disasm); exp fallback otherwise.
__device__ __forceinline__ float fast_tanh(float u) {
#if __has_builtin(__builtin_amdgcn_tanhf)
    return __builtin_amdgcn_tanhf(u);
#else
    float e = __expf(2.0f * u);
    return 1.0f - 2.0f / (e + 1.0f);
#endif
}

__device__ __forceinline__ float fast_sigmoid(float v) {
    return fmaf(0.5f, fast_tanh(0.5f * v), 0.5f);
}

// ---------------------------------------------------------------------------
// prep_rd: s2[r,d] = 0.5*kappa*tanh(ineq), w[r,d] = sigmoid(mask)*tanh(e_sign)
// ---------------------------------------------------------------------------
__global__ void prep_rd_kernel(const float* __restrict__ ineq,
                               const float* __restrict__ esign,
                               const float* __restrict__ mlogit,
                               const float* __restrict__ logk,
                               float* __restrict__ s2,
                               float* __restrict__ w) {
    int i = blockIdx.x * blockDim.x + threadIdx.x;
    if (i >= RDN) return;
    float kappa = __expf(logk[0]);
    kappa = fminf(fmaxf(kappa, 0.5f), 50.0f);
    s2[i] = 0.5f * kappa * fast_tanh(ineq[i]);
    w[i]  = fast_sigmoid(mlogit[i]) * fast_tanh(esign[i]);
}

// ---------------------------------------------------------------------------
// prep_bpack: softmax(group_logits, axis=G) packed into WMMA B-operand layout.
// B operand (16-bit, 32x16): lane L holds column n = L%16; element e holds
// k = 8*(L>=16) + e for e<8, 16 + 8*(L>=16) + (e-8) for e>=8.  Columns n>=4 are zero padding.
// ---------------------------------------------------------------------------
__global__ void prep_bpack_kernel(const float* __restrict__ gl,
                                  _Float16* __restrict__ bpack) {
    int t = threadIdx.x;            // 0..511  == chunk*32 + lane
    if (t >= 512) return;
    int lane = t & 31;
    int chunk = t >> 5;
    int n  = lane & 15;
    int hs = lane >> 4;
#pragma unroll
    for (int e = 0; e < 16; ++e) {
        int k = (e < 8) ? (e + 8 * hs) : (16 + 8 * hs + (e - 8));
        int d = chunk * 32 + k;
        float val = 0.0f;
        if (n < GN) {
            float g0 = gl[0 * DN + d], g1 = gl[1 * DN + d];
            float g2 = gl[2 * DN + d], g3 = gl[3 * DN + d];
            float mx = fmaxf(fmaxf(g0, g1), fmaxf(g2, g3));
            float e0 = __expf(g0 - mx), e1 = __expf(g1 - mx);
            float e2 = __expf(g2 - mx), e3 = __expf(g3 - mx);
            float s = e0 + e1 + e2 + e3;
            float sel = (n == 0) ? e0 : ((n == 1) ? e1 : ((n == 2) ? e2 : e3));
            val = sel / s;
        }
        bpack[t * 16 + e] = (_Float16)val;
    }
}

// A-operand d-pattern loader: two contiguous 8-float runs at p+0 and p+16.
__device__ __forceinline__ void load16(const float* __restrict__ p, float v[16]) {
    const float4 a = *(const float4*)(p + 0);
    const float4 b = *(const float4*)(p + 4);
    const float4 c = *(const float4*)(p + 16);
    const float4 d = *(const float4*)(p + 20);
    v[0] = a.x;  v[1] = a.y;  v[2] = a.z;  v[3] = a.w;
    v[4] = b.x;  v[5] = b.y;  v[6] = b.z;  v[7] = b.w;
    v[8] = c.x;  v[9] = c.y;  v[10] = c.z; v[11] = c.w;
    v[12] = d.x; v[13] = d.y; v[14] = d.z; v[15] = d.w;
}

// Stage the contiguous 16x512 f32 x-tile (32 KB) into padded LDS rows using
// gfx1250 async global->LDS b128 copies (ASYNCcnt), fallback: load+store.
__device__ __forceinline__ void stage_x_tile(const float* __restrict__ gsrc,
                                             float* __restrict__ lx, int tid) {
#pragma unroll
    for (int it = 0; it < 8; ++it) {
        int idx = it * 256 + tid;          // float4 index, 0..2047
        int row = idx >> 7;                // 128 float4 per row
        int c4  = idx & 127;
        const float* g = gsrc + (size_t)idx * 4;
        float* l = lx + row * LXS + c4 * 4;
#if defined(__gfx1250__)
        asm volatile("global_load_async_to_lds_b128 %0, %1, off"
                     :: "v"((lds_void*)l), "v"(g) : "memory");
#else
        *(float4*)l = *(const float4*)g;
#endif
    }
#if defined(__gfx1250__)
    asm volatile("s_wait_asynccnt 0x0" ::: "memory");
#endif
}

// ---------------------------------------------------------------------------
// fused kernel: per wave, one 16-b tile x one r.  16 chained WMMAs over D.
// grid = (B/16, 8), block = 256 (8 waves); each wave walks 4 r values.
// x tile staged in LDS once per block (async-to-LDS); th/s2/w stream from L2.
// ---------------------------------------------------------------------------
__global__ void __launch_bounds__(256)
fused_kernel(const float* __restrict__ x,  const float* __restrict__ th,
             const float* __restrict__ tg, const float* __restrict__ gml,
             const float* __restrict__ kfrac, const float* __restrict__ headw,
             const float* __restrict__ s2, const float* __restrict__ w,
             const _Float16* __restrict__ bpack,
             float* __restrict__ partial) {
    const int lane  = threadIdx.x & 31;
    const int wave  = threadIdx.x >> 5;
    const int m     = lane & 15;   // A-operand row / D-operand column index
    const int hs    = lane >> 4;   // lane half select
    const int btile = blockIdx.x;
    const int r0    = blockIdx.y * 32;

    __shared__ float lx[16 * LXS];     // 33 KB padded x tile
    __shared__ float red[8][16];

    stage_x_tile(x + (size_t)btile * 16 * DN, lx, threadIdx.x);
    __syncthreads();

    float yacc[8];
#pragma unroll
    for (int j = 0; j < 8; ++j) yacc[j] = 0.0f;

    for (int rr = wave; rr < 32; rr += 8) {
        const int r = r0 + rr;
        const float* xr  = lx + m * LXS;          // this lane's b row (LDS)
        const float* thr = th + (size_t)r * DN;
        const float* s2r = s2 + (size_t)r * DN;
        const float* wr  = w  + (size_t)r * DN;

        v8f acc = {};
        for (int chunk = 0; chunk < 16; ++chunk) {
            const int dbase = chunk * 32 + hs * 8;
            float xv[16], tv[16], sv[16], wv[16];
            load16(xr  + dbase, xv);
            load16(thr + dbase, tv);
            load16(s2r + dbase, sv);
            load16(wr  + dbase, wv);

            v16h amat;
#pragma unroll
            for (int e = 0; e < 16; ++e) {
                float u = sv[e] * (xv[e] - tv[e]);          // kappa*ineq*(x-th)/2
                amat[e] = (_Float16)(wv[e] * fast_tanh(u)); // m*e_sign*(2c-1)
            }
            v16h bmat = *(const v16h*)(bpack + (chunk * 32 + lane) * 16);

            acc = __builtin_amdgcn_wmma_f32_16x16x32_f16(
                false, amat, false, bmat, (short)0, acc, false, false);
        }

        // --- epilogue: acc[j] = eg[b = btile*16 + j + 8*hs][g = n] ----------
        // branch-free: clamped index + multiplicative mask (no exec toggling)
        const int n  = m;
        const int nc = n & (GN - 1);
        const float gsel = (n < GN) ? 1.0f : 0.0f;
        const float gmv  = fast_sigmoid(gml[r * GN + nc]) * gsel;
        const float tgv  = tg[r * GN + nc];
        float en = 1e-6f;
#pragma unroll
        for (int g = 0; g < GN; ++g) en += fast_sigmoid(gml[r * GN + g]);
        const float kk = fast_sigmoid(kfrac[r]) * en;
        const float hw = headw[r];

#pragma unroll
        for (int j = 0; j < 8; ++j) {
            float pg = gmv * fast_sigmoid(6.0f * (acc[j] - tgv));
            pg += __shfl_xor(pg, 1, 32);   // reduce over g in lanes n=0..3
            pg += __shfl_xor(pg, 2, 32);
            float z = fast_sigmoid(8.0f * (pg - kk));
            yacc[j] += z * hw;             // valid where n == 0
        }
    }

    if (m == 0) {
#pragma unroll
        for (int j = 0; j < 8; ++j) red[wave][hs * 8 + j] = yacc[j];
    }
    __syncthreads();
    if (threadIdx.x < 16) {
        float s = 0.0f;
#pragma unroll
        for (int wv = 0; wv < 8; ++wv) s += red[wv][threadIdx.x];
        partial[(size_t)blockIdx.y * BN + btile * 16 + threadIdx.x] = s;
    }
}

// ---------------------------------------------------------------------------
// finalize: out[b] = head_b + sum over the 8 r-partitions
// ---------------------------------------------------------------------------
__global__ void finalize_kernel(const float* __restrict__ partial,
                                const float* __restrict__ headb,
                                float* __restrict__ out) {
    int b = blockIdx.x * blockDim.x + threadIdx.x;
    if (b >= BN) return;
    float s = headb[0];
#pragma unroll
    for (int p = 0; p < 8; ++p) s += partial[(size_t)p * BN + b];
    out[b] = s;
}

extern "C" void kernel_launch(void* const* d_in, const int* in_sizes, int n_in,
                              void* d_out, int out_size, void* d_ws, size_t ws_size,
                              hipStream_t stream) {
    const float* x      = (const float*)d_in[0];   // (B,D)
    const float* th     = (const float*)d_in[1];   // (R,D)
    const float* ineq   = (const float*)d_in[2];   // (R,D)
    const float* esign  = (const float*)d_in[3];   // (R,D)
    const float* mlogit = (const float*)d_in[4];   // (R,D)
    const float* logk   = (const float*)d_in[5];   // scalar
    const float* glog   = (const float*)d_in[6];   // (G,D)
    const float* tg     = (const float*)d_in[7];   // (R,G)
    const float* gml    = (const float*)d_in[8];   // (R,G)
    const float* kfrac  = (const float*)d_in[9];   // (R,)
    const float* headw  = (const float*)d_in[10];  // (1,R)
    const float* headb  = (const float*)d_in[11];  // (1,)
    float* out = (float*)d_out;

    // workspace layout
    float*    s2      = (float*)d_ws;                       // R*D f32
    float*    w       = s2 + RDN;                           // R*D f32
    _Float16* bpack   = (_Float16*)(w + RDN);               // 16*32*16 f16
    float*    partial = (float*)(bpack + 16 * 32 * 16);     // 8*B f32
    (void)in_sizes; (void)n_in; (void)out_size; (void)ws_size;

    prep_rd_kernel<<<(RDN + 255) / 256, 256, 0, stream>>>(ineq, esign, mlogit,
                                                          logk, s2, w);
    prep_bpack_kernel<<<1, 512, 0, stream>>>(glog, bpack);
    fused_kernel<<<dim3(BN / 16, 8), 256, 0, stream>>>(x, th, tg, gml, kfrac,
                                                       headw, s2, w, bpack,
                                                       partial);
    finalize_kernel<<<(BN + 255) / 256, 256, 0, stream>>>(partial, headb, out);
}
